// Block_42021960024367
// MI455X (gfx1250) — compile-verified, matrix operand order checked by reading, not run on hardware
//
#include <hip/hip_runtime.h>
#include <hip/hip_bf16.h>

typedef __attribute__((ext_vector_type(8)))  __bf16 v8bf;
typedef __attribute__((ext_vector_type(16))) __bf16 v16bf;
typedef __attribute__((ext_vector_type(8)))  float  v8f;
typedef unsigned int u32x4 __attribute__((ext_vector_type(4)));
typedef int          i32x4 __attribute__((ext_vector_type(4)));
typedef int          i32x8 __attribute__((ext_vector_type(8)));

#define DIM 1024
#define HEADS 16
#define HD 64
#define HIDDEN 4096
#define BATCH 16
#define SEQ 1024
#define ROWS (BATCH * SEQ)   // 16384 tokens

// ---------------------------------------------------------------------------
// Fragment loader: CDNA5 16-bit A/B layout for v_wmma_*_16x16x32.
// For lane = half*16 + m, vector element j holds K = koff + 8*half + j (j<8)
// and K = koff + 16 + 8*half + (j-8) (j>=8)  -> two contiguous 16B loads.
// ---------------------------------------------------------------------------
__device__ __forceinline__ v16bf frag_ld(const __bf16* rowp, int half, int koff) {
    const v8bf lo = *(const v8bf*)(rowp + koff + 8 * half);
    const v8bf hi = *(const v8bf*)(rowp + koff + 16 + 8 * half);
    v16bf f;
#pragma unroll
    for (int i = 0; i < 8; ++i) { f[i] = lo[i]; f[i + 8] = hi[i]; }
    return f;
}

__device__ __forceinline__ v8f zero8() {
    v8f z = {0.f, 0.f, 0.f, 0.f, 0.f, 0.f, 0.f, 0.f};
    return z;
}

__device__ __forceinline__ float gelu_exact(float v) {
    return 0.5f * v * (1.0f + erff(v * 0.70710678118654752f));
}

// ---------------------------------------------------------------------------
// TDM: issue a 2D tensor_load_to_lds of a (tile_d1 rows x tile_d0 elems)
// bf16 tile with row stride `stride_d0` (elements) into LDS at lds_addr.
// LDS padding via D#: after every 16 DWORDs (64B = one 32-elem bf16 row),
// skip 4 DWORDs (16B) -> LDS row stride 80B == [.][40] __bf16 layout.
// Descriptor packing per CDNA5 ISA section 8.3/8.4 (D# groups 0/1).
// ---------------------------------------------------------------------------
__device__ __forceinline__ void tdm_load_tile_2d_bf16(
    unsigned int lds_addr, unsigned long long gaddr,
    unsigned int tile_d0, unsigned int tile_d1,
    unsigned long long stride_d0) {
    const unsigned int tensor_d0 = 1u << 30;   // never clamp (tiles in-bounds)
    const unsigned int tensor_d1 = 1u << 20;

    u32x4 g0;
    g0[0] = 1u;                                            // count=1, user D#
    g0[1] = lds_addr;                                      // [63:32] lds_addr
    g0[2] = (unsigned int)gaddr;                           // [95:64]
    g0[3] = (unsigned int)(gaddr >> 32) | (2u << 30);      // [120:96] + type=2

    i32x8 g1;
    // wg_mask=0 | data_size=1(2B)<<16 | pad_enable<<20 |
    // pad_interval=3(16 DW)<<22 | pad_amount=3(4 DW)<<25
    g1[0] = (1 << 16) | (1 << 20) | (3 << 22) | (3 << 25);
    g1[1] = (int)((tensor_d0 & 0xFFFFu) << 16);            // dim0[15:0] @63:48
    g1[2] = (int)((tensor_d0 >> 16) | ((tensor_d1 & 0xFFFFu) << 16));
    g1[3] = (int)((tensor_d1 >> 16) | (tile_d0 << 16));    // tile_dim0 @127:112
    g1[4] = (int)(tile_d1 & 0xFFFFu);                      // tile_dim1; tile_dim2=0
    g1[5] = (int)(unsigned int)stride_d0;                  // dim0_stride[31:0]
    g1[6] = (int)(unsigned int)((stride_d0 >> 32) & 0xFFFFu);
    g1[7] = 0;

    i32x4 gz = {0, 0, 0, 0};
#if __clang_major__ >= 23
    i32x8 gz8 = {0, 0, 0, 0, 0, 0, 0, 0};
    __builtin_amdgcn_tensor_load_to_lds(g0, g1, gz, gz, gz8, 0);
#else
    __builtin_amdgcn_tensor_load_to_lds(g0, g1, gz, gz, 0);
#endif
}

// ---------------------------------------------------------------------------
// Weight transpose+convert: w[K][N] f32 -> wT[N][K] bf16 (GEMM B operand)
// ---------------------------------------------------------------------------
__global__ void wt_convert_kernel(const float* __restrict__ w,
                                  __bf16* __restrict__ wT, int K, int N) {
    size_t idx = (size_t)blockIdx.x * blockDim.x + threadIdx.x;
    int nrow = (int)(idx / (size_t)K);
    int k    = (int)(idx % (size_t)K);
    if (nrow < N) wT[idx] = (__bf16)w[(size_t)k * N + nrow];
}

// ---------------------------------------------------------------------------
// LayerNorm over 1024 f32, emit bf16 (feeds WMMA GEMMs). One row per block.
// ---------------------------------------------------------------------------
__global__ __launch_bounds__(256) void layernorm_bf16_kernel(
    const float* __restrict__ x, const float* __restrict__ g,
    const float* __restrict__ b, __bf16* __restrict__ out) {
    const int row = blockIdx.x;
    const int tid = threadIdx.x;
    const float4 v = ((const float4*)(x + (size_t)row * DIM))[tid];

    float s  = v.x + v.y + v.z + v.w;
    float ss = v.x * v.x + v.y * v.y + v.z * v.z + v.w * v.w;
#pragma unroll
    for (int mask = 1; mask < 32; mask <<= 1) {
        s  += __shfl_xor(s,  mask, 32);
        ss += __shfl_xor(ss, mask, 32);
    }
    __shared__ float red[16];
    const int wid = tid >> 5, lane = tid & 31;
    if (lane == 0) { red[wid] = s; red[8 + wid] = ss; }
    __syncthreads();
    s = 0.f; ss = 0.f;
#pragma unroll
    for (int w = 0; w < 8; ++w) { s += red[w]; ss += red[8 + w]; }
    const float mu  = s * (1.0f / DIM);
    const float var = ss * (1.0f / DIM) - mu * mu;
    const float rs  = rsqrtf(var + 1e-5f);

    const float4 gg = ((const float4*)g)[tid];
    const float4 bb = ((const float4*)b)[tid];
    __bf16* orow = out + (size_t)row * DIM + tid * 4;
    orow[0] = (__bf16)((v.x - mu) * rs * gg.x + bb.x);
    orow[1] = (__bf16)((v.y - mu) * rs * gg.y + bb.y);
    orow[2] = (__bf16)((v.z - mu) * rs * gg.z + bb.z);
    orow[3] = (__bf16)((v.w - mu) * rs * gg.w + bb.w);
}

// ---------------------------------------------------------------------------
// Tiled BF16 WMMA GEMM with double-buffered TDM staging.
// C[M][N] = A[M][K] * Bt[N][K]^T (+ epilogue)
// 256 threads = 8 waves (2x4); block tile 128x128, K-step 32.
// Wave 0 issues tensor_load_to_lds for tile k+1 while all waves run WMMAs
// on tile k; completion via TENSORcnt. One barrier per K-step.
// EPI: 0 = bias -> f32   1 = bias+GELU -> bf16   2 = bias+residual -> f32
// ---------------------------------------------------------------------------
template <int EPI>
__global__ __launch_bounds__(256) void gemm_bf16_kernel(
    const __bf16* __restrict__ A, const __bf16* __restrict__ Bt,
    const float* __restrict__ bias, const float* __restrict__ res,
    float* __restrict__ Cf, __bf16* __restrict__ Cb, int M, int N, int K) {
    __shared__ __align__(16) __bf16 As[2][128][40];   // row stride 80B (TDM pad)
    __shared__ __align__(16) __bf16 Bs[2][128][40];

    const int tid  = threadIdx.x;
    const int lane = tid & 31, wid = tid >> 5;
    const int half = lane >> 4, ln16 = lane & 15;
    const int wm = wid & 1, wn = wid >> 1;           // wave grid 2 (M) x 4 (N)
    const int rowBase = blockIdx.y * 128;
    const int colBase = blockIdx.x * 128;

    v8f acc[4][2];
#pragma unroll
    for (int i = 0; i < 4; ++i)
#pragma unroll
        for (int j = 0; j < 2; ++j) acc[i][j] = zero8();

    const unsigned long long aBase =
        (unsigned long long)(uintptr_t)(A + (size_t)rowBase * K);
    const unsigned long long bBase =
        (unsigned long long)(uintptr_t)(Bt + (size_t)colBase * K);
    const unsigned int ldsA0 = (unsigned int)(uintptr_t)&As[0][0][0];
    const unsigned int ldsA1 = (unsigned int)(uintptr_t)&As[1][0][0];
    const unsigned int ldsB0 = (unsigned int)(uintptr_t)&Bs[0][0][0];
    const unsigned int ldsB1 = (unsigned int)(uintptr_t)&Bs[1][0][0];

    // Prime the pipeline: DMA first K-tile into buffer 0.
    if (wid == 0) {
        tdm_load_tile_2d_bf16(ldsA0, aBase, 32, 128, (unsigned long long)K);
        tdm_load_tile_2d_bf16(ldsB0, bBase, 32, 128, (unsigned long long)K);
    }

    int buf = 0;
    for (int k0 = 0; k0 < K; k0 += 32) {
        if (wid == 0) __builtin_amdgcn_s_wait_tensorcnt(0);
        __syncthreads();   // tile `buf` visible; tile `buf^1` fully consumed

        if (wid == 0 && k0 + 32 < K) {   // DMA next tile while we compute
            const unsigned long long goff = (unsigned long long)(k0 + 32) * 2;
            tdm_load_tile_2d_bf16(buf ? ldsA0 : ldsA1, aBase + goff, 32, 128,
                                  (unsigned long long)K);
            tdm_load_tile_2d_bf16(buf ? ldsB0 : ldsB1, bBase + goff, 32, 128,
                                  (unsigned long long)K);
        }

        v16bf af[4], bf[2];
#pragma unroll
        for (int i = 0; i < 4; ++i)
            af[i] = frag_ld(&As[buf][wm * 64 + i * 16 + ln16][0], half, 0);
#pragma unroll
        for (int j = 0; j < 2; ++j)
            bf[j] = frag_ld(&Bs[buf][wn * 32 + j * 16 + ln16][0], half, 0);
#pragma unroll
        for (int i = 0; i < 4; ++i)
#pragma unroll
            for (int j = 0; j < 2; ++j)
                acc[i][j] = __builtin_amdgcn_wmma_f32_16x16x32_bf16(
                    false, af[i], false, bf[j], (short)0, acc[i][j], false, false);
        buf ^= 1;
    }

    // Epilogue. C layout: lane holds col n = ln16; element r -> row r + 8*half.
#pragma unroll
    for (int j = 0; j < 2; ++j) {
        const int col = colBase + wn * 32 + j * 16 + ln16;
        const float bv = bias[col];
#pragma unroll
        for (int i = 0; i < 4; ++i) {
            const int row0 = rowBase + wm * 64 + i * 16 + 8 * half;
#pragma unroll
            for (int r = 0; r < 8; ++r) {
                const size_t off = (size_t)(row0 + r) * N + col;
                float v = acc[i][j][r] + bv;
                if (EPI == 1) {
                    Cb[off] = (__bf16)gelu_exact(v);
                } else if (EPI == 2) {
                    Cf[off] = v + res[off];
                } else {
                    Cf[off] = v;
                }
            }
        }
    }
}

// ---------------------------------------------------------------------------
// Split qkv f32 [B,N,3,H,hd] -> q (pre-scaled by hd^-0.5), k bf16 [B,H,N,hd];
// v transposed bf16 [B,H,hd,N]
// ---------------------------------------------------------------------------
__global__ void qkv_split_kernel(const float* __restrict__ qkv,
                                 __bf16* __restrict__ q, __bf16* __restrict__ k,
                                 __bf16* __restrict__ vT) {
    const size_t idx = (size_t)blockIdx.x * blockDim.x + threadIdx.x;
    const int col = (int)(idx % (3 * DIM));
    const size_t row = idx / (3 * DIM);          // row = b*SEQ + n
    const int b = (int)(row >> 10), n = (int)(row & 1023);
    const int g = col >> 10, rem = col & 1023;
    const int h = rem >> 6, d = rem & 63;
    const float v = qkv[idx];
    const size_t bh = (size_t)(b * HEADS + h);
    if (g == 0)      q[(bh * SEQ + n) * HD + d] = (__bf16)(v * 0.125f);
    else if (g == 1) k[(bh * SEQ + n) * HD + d] = (__bf16)v;
    else             vT[(bh * HD + d) * SEQ + n] = (__bf16)v;
}

// ---------------------------------------------------------------------------
// Flash attention: one (b,h) x 64-query block per workgroup (4 waves,
// 16 queries/wave). Key chunks of 64. WMMA for S=QK^T and O=PV, online
// softmax in f32 (softmax scale pre-folded into q), P transposed C->A layout
// through wave-private LDS. Writes ctx in [B,N,H*hd] bf16 (proj GEMM A).
// ---------------------------------------------------------------------------
__global__ __launch_bounds__(128) void flash_attn_kernel(
    const __bf16* __restrict__ Q, const __bf16* __restrict__ Km,
    const __bf16* __restrict__ VT, __bf16* __restrict__ ctx) {
    const int bh = blockIdx.y;                  // b*HEADS + h
    const int b = bh >> 4, h = bh & 15;
    const int qblk = blockIdx.x;                // 64 queries each
    const int tid = threadIdx.x, lane = tid & 31, wv = tid >> 5;
    const int half = lane >> 4, ln16 = lane & 15;

    __shared__ __align__(16) __bf16 Plds[4][16][72];

    const size_t base = (size_t)bh * SEQ * HD;  // same for Q,K,VT
    const int qrow = qblk * 64 + wv * 16 + ln16;
    const __bf16* qp = Q + base + (size_t)qrow * HD;
    const v16bf qf0 = frag_ld(qp, half, 0);     // d = 0..31
    const v16bf qf1 = frag_ld(qp, half, 32);    // d = 32..63

    float mrow[8], lrow[8];
    v8f acco[4];
#pragma unroll
    for (int r = 0; r < 8; ++r) { mrow[r] = -3.0e38f; lrow[r] = 0.f; }
#pragma unroll
    for (int dt = 0; dt < 4; ++dt) acco[dt] = zero8();

    for (int kc = 0; kc < SEQ; kc += 64) {
        // ---- S = Q K^T over this 64-key chunk: 4 tiles of 16x16 ----
        v8f s[4];
#pragma unroll
        for (int t = 0; t < 4; ++t) {
            const __bf16* kp = Km + base + (size_t)(kc + t * 16 + ln16) * HD;
            const v16bf kf0 = frag_ld(kp, half, 0);
            const v16bf kf1 = frag_ld(kp, half, 32);
            s[t] = zero8();
            s[t] = __builtin_amdgcn_wmma_f32_16x16x32_bf16(
                false, qf0, false, kf0, (short)0, s[t], false, false);
            s[t] = __builtin_amdgcn_wmma_f32_16x16x32_bf16(
                false, qf1, false, kf1, (short)0, s[t], false, false);
        }
        // ---- online softmax (rows r+8*half spread over 16 lanes) ----
        float vmax[8];
#pragma unroll
        for (int r = 0; r < 8; ++r) {
            float mx = s[0][r];
#pragma unroll
            for (int t = 1; t < 4; ++t) mx = fmaxf(mx, s[t][r]);
#pragma unroll
            for (int mask = 1; mask < 16; mask <<= 1)
                mx = fmaxf(mx, __shfl_xor(mx, mask, 32));
            vmax[r] = mx;
        }
        float alpha[8];
#pragma unroll
        for (int r = 0; r < 8; ++r) {
            const float mnew = fmaxf(mrow[r], vmax[r]);
            alpha[r] = __expf(mrow[r] - mnew);
            mrow[r] = mnew;
        }
#pragma unroll
        for (int t = 0; t < 4; ++t)
#pragma unroll
            for (int r = 0; r < 8; ++r)
                s[t][r] = __expf(s[t][r] - mrow[r]);
#pragma unroll
        for (int r = 0; r < 8; ++r) {
            float rs = s[0][r] + s[1][r] + s[2][r] + s[3][r];
#pragma unroll
            for (int mask = 1; mask < 16; mask <<= 1)
                rs += __shfl_xor(rs, mask, 32);
            lrow[r] = lrow[r] * alpha[r] + rs;
        }
#pragma unroll
        for (int dt = 0; dt < 4; ++dt)
#pragma unroll
            for (int r = 0; r < 8; ++r) acco[dt][r] *= alpha[r];

        // ---- transpose P (C layout -> A layout) via wave-private LDS ----
#pragma unroll
        for (int t = 0; t < 4; ++t)
#pragma unroll
            for (int r = 0; r < 8; ++r)
                Plds[wv][r + 8 * half][t * 16 + ln16] = (__bf16)s[t][r];
        asm volatile("s_wait_dscnt 0" ::: "memory");
        const v16bf pf0 = frag_ld(&Plds[wv][ln16][0], half, 0);
        const v16bf pf1 = frag_ld(&Plds[wv][ln16][0], half, 32);

        // ---- O += P V  (VT rows = d, contiguous keys) ----
#pragma unroll
        for (int dt = 0; dt < 4; ++dt) {
            const __bf16* vp = VT + base + (size_t)(dt * 16 + ln16) * SEQ + kc;
            const v16bf vf0 = frag_ld(vp, half, 0);
            const v16bf vf1 = frag_ld(vp, half, 32);
            acco[dt] = __builtin_amdgcn_wmma_f32_16x16x32_bf16(
                false, pf0, false, vf0, (short)0, acco[dt], false, false);
            acco[dt] = __builtin_amdgcn_wmma_f32_16x16x32_bf16(
                false, pf1, false, vf1, (short)0, acco[dt], false, false);
        }
    }

    // ---- write ctx[b, n, h*64 + d] bf16 ----
#pragma unroll
    for (int dt = 0; dt < 4; ++dt)
#pragma unroll
        for (int r = 0; r < 8; ++r) {
            const int row = qblk * 64 + wv * 16 + r + 8 * half;
            const float ov = acco[dt][r] / lrow[r];
            ctx[((size_t)(b * SEQ + row)) * DIM + h * HD + dt * 16 + ln16] =
                (__bf16)ov;
        }
}

// ---------------------------------------------------------------------------
// Host launcher
// ---------------------------------------------------------------------------
extern "C" void kernel_launch(void* const* d_in, const int* in_sizes, int n_in,
                              void* d_out, int out_size, void* d_ws, size_t ws_size,
                              hipStream_t stream) {
    const float* x      = (const float*)d_in[0];
    const float* ln1_g  = (const float*)d_in[1];
    const float* ln1_b  = (const float*)d_in[2];
    const float* w_qkv  = (const float*)d_in[3];
    const float* b_qkv  = (const float*)d_in[4];
    const float* w_proj = (const float*)d_in[5];
    const float* b_proj = (const float*)d_in[6];
    const float* ln2_g  = (const float*)d_in[7];
    const float* ln2_b  = (const float*)d_in[8];
    const float* w_fc1  = (const float*)d_in[9];
    const float* b_fc1  = (const float*)d_in[10];
    const float* w_fc2  = (const float*)d_in[11];
    const float* b_fc2  = (const float*)d_in[12];
    float* out = (float*)d_out;

    char* ws = (char*)d_ws;
    size_t o = 0;
    auto carve = [&](size_t bytes) -> char* {
        char* p = ws + o;
        o += (bytes + 255) & ~(size_t)255;
        return p;
    };
    __bf16* wT_qkv  = (__bf16*)carve((size_t)3 * DIM * DIM * 2);
    __bf16* wT_proj = (__bf16*)carve((size_t)DIM * DIM * 2);
    __bf16* wT_fc1  = (__bf16*)carve((size_t)HIDDEN * DIM * 2);
    __bf16* wT_fc2  = (__bf16*)carve((size_t)DIM * HIDDEN * 2);
    __bf16* h1      = (__bf16*)carve((size_t)ROWS * DIM * 2);    // aliased as h2
    float*  qkvf    = (float*)carve((size_t)ROWS * 3 * DIM * 4); // aliased as mb
    __bf16* qb      = (__bf16*)carve((size_t)ROWS * DIM * 2);
    __bf16* kb      = (__bf16*)carve((size_t)ROWS * DIM * 2);
    __bf16* vTb     = (__bf16*)carve((size_t)ROWS * DIM * 2);
    __bf16* ctx     = (__bf16*)carve((size_t)ROWS * DIM * 2);
    float*  x1      = (float*)carve((size_t)ROWS * DIM * 4);
    __bf16* h2 = h1;                 // h1 dead after QKV GEMM
    __bf16* mb = (__bf16*)qkvf;      // qkv f32 dead after split (134MB <= 201MB)

    // 1) weights -> bf16 [N][K]
    wt_convert_kernel<<<(3 * DIM * DIM) / 256, 256, 0, stream>>>(w_qkv, wT_qkv, DIM, 3 * DIM);
    wt_convert_kernel<<<(DIM * DIM) / 256, 256, 0, stream>>>(w_proj, wT_proj, DIM, DIM);
    wt_convert_kernel<<<(HIDDEN * DIM) / 256, 256, 0, stream>>>(w_fc1, wT_fc1, DIM, HIDDEN);
    wt_convert_kernel<<<(DIM * HIDDEN) / 256, 256, 0, stream>>>(w_fc2, wT_fc2, HIDDEN, DIM);

    // 2) LN1
    layernorm_bf16_kernel<<<ROWS, 256, 0, stream>>>(x, ln1_g, ln1_b, h1);

    // 3) QKV GEMM: [16384,1024] x [1024,3072]
    gemm_bf16_kernel<0><<<dim3((3 * DIM) / 128, ROWS / 128), 256, 0, stream>>>(
        h1, wT_qkv, b_qkv, nullptr, qkvf, nullptr, ROWS, 3 * DIM, DIM);

    // 4) split into q(scaled)/k [B,H,N,hd] and vT [B,H,hd,N]
    qkv_split_kernel<<<((size_t)ROWS * 3 * DIM) / 256, 256, 0, stream>>>(qkvf, qb, kb, vTb);

    // 5) attention
    flash_attn_kernel<<<dim3(SEQ / 64, BATCH * HEADS), 128, 0, stream>>>(qb, kb, vTb, ctx);

    // 6) proj GEMM + residual -> x1 (f32)
    gemm_bf16_kernel<2><<<dim3(DIM / 128, ROWS / 128), 256, 0, stream>>>(
        ctx, wT_proj, b_proj, x, x1, nullptr, ROWS, DIM, DIM);

    // 7) LN2
    layernorm_bf16_kernel<<<ROWS, 256, 0, stream>>>(x1, ln2_g, ln2_b, h2);

    // 8) FC1 GEMM + GELU -> bf16
    gemm_bf16_kernel<1><<<dim3(HIDDEN / 128, ROWS / 128), 256, 0, stream>>>(
        h2, wT_fc1, b_fc1, nullptr, nullptr, mb, ROWS, HIDDEN, DIM);

    // 9) FC2 GEMM + residual -> out
    gemm_bf16_kernel<2><<<dim3(DIM / 128, ROWS / 128), 256, 0, stream>>>(
        mb, wT_fc2, b_fc2, x1, out, nullptr, ROWS, DIM, HIDDEN);
}
